// SimpleFPCNN3_86414741995918
// MI455X (gfx1250) — matrix operand order repaired and checked by reading, not compiled, not run on hardware
//
#include <hip/hip_runtime.h>

// ---------------------------------------------------------------------------
// CDNA5 (gfx1250) implementation of SimpleFPCNN3 denoiser.
// All 3x3 convolutions run as implicit-GEMM on v_wmma_f32_16x16x32_f16.
// A/B fragments are fed by contiguous ds_load_b128 from a K-major im2col
// tile staged in LDS (fp32 -> fp16). K-loop fully unrolled per Cin template.
// Bilateral / pool / upsample are VALU fp32. Working set is L2-resident.
// ---------------------------------------------------------------------------

typedef __attribute__((ext_vector_type(16))) _Float16 v16h;
typedef __attribute__((ext_vector_type(8)))  _Float16 v8h;
typedef __attribute__((ext_vector_type(8)))  float    v8f;

#define TILE_W 64

// ---------------------------------------------------------------------------
// Generic 3x3 conv (stride 1, pad 1) via WMMA, templated on Cin.
//   in  : [Cin][H][W]  fp32
//   wgt : [Cout][Cin][3][3] fp32
//   out : [Cout][H][W] fp32 (bias added, optional ReLU)
// Block = 128 threads = 4 waves. Each wave: 16 pixels x 16 out-channels tile.
// LDS: im2col [64][Krow] halves (K-major)  +  bmatT [16][Krow] halves.
// A fragment = 2x ds_load_b128 (K runs [kt+8h, +8) and [kt+16+8h, +8)).
// B fragment = 2x ds_load_b128 (K runs [kt+8h, +8) and [kt+16+8h, +8)).
// ---------------------------------------------------------------------------
template<int CIN>
__global__ __launch_bounds__(128)
void conv3x3_wmma(const float* __restrict__ in, const float* __restrict__ wgt,
                  const float* __restrict__ bias, float* __restrict__ out,
                  int Cout, int H, int W, int relu)
{
    constexpr int K    = CIN * 9;
    constexpr int Kpad = (K + 31) & ~31;
    constexpr int Krow = Kpad + 8;     // +16B pad: bank spread, keeps 16B align

    extern __shared__ __align__(16) char smem[];
    _Float16* im2col = (_Float16*)smem;          // [64][Krow]
    _Float16* bmatT  = im2col + 64 * Krow;       // [16][Krow]

    const int tid  = threadIdx.x;
    const int lane = tid & 31;
    const int wv   = tid >> 5;            // wave id 0..3
    const int x0   = blockIdx.x * TILE_W;
    const int y    = blockIdx.y;
    const int oc0  = blockIdx.z * 16;
    const long HW  = (long)H * W;

    // ---- stage im2col tile: m minor => coalesced global reads ------------
    for (int i = tid; i < 64 * Kpad; i += 128) {
        int m = i & 63;
        int k = i >> 6;
        float v = 0.0f;
        if (k < K) {
            int c = k / 9, t = k % 9;
            int gy = y + (t / 3) - 1;
            int gx = x0 + m + (t % 3) - 1;
            if ((unsigned)gy < (unsigned)H && (unsigned)gx < (unsigned)W)
                v = in[(long)c * HW + (long)gy * W + gx];
        }
        im2col[m * Krow + k] = (_Float16)v;
    }
    // ---- stage B transposed [n][k], zero padded --------------------------
    for (int i = tid; i < 16 * Kpad; i += 128) {
        int n = i / Kpad, k = i % Kpad;
        int oc = oc0 + n;
        float v = 0.0f;
        if (k < K && oc < Cout)
            v = wgt[(long)oc * K + k];    // [oc][c][ky][kx] flat == [oc][k]
        bmatT[n * Krow + k] = (_Float16)v;
    }
    __syncthreads();

    const int ln15  = lane & 15;          // A row m (low), B col n, C col n
    const int khalf = lane >> 4;          // 0 or 1
    const int oc    = oc0 + ln15;
    float bv = (oc < Cout) ? bias[oc] : 0.0f;
    v8f acc;
#pragma unroll
    for (int i = 0; i < 8; ++i) acc[i] = bv;

    const _Float16* arow = im2col + (wv * 16 + ln15) * Krow;
    const _Float16* brow = bmatT + ln15 * Krow;

#pragma unroll
    for (int kt = 0; kt < Kpad; kt += 32) {
        v8h a0 = *(const v8h*)(arow + kt + khalf * 8);
        v8h a1 = *(const v8h*)(arow + kt + 16 + khalf * 8);
        v8h b0 = *(const v8h*)(brow + kt + khalf * 8);
        v8h b1 = *(const v8h*)(brow + kt + 16 + khalf * 8);
        v16h a = __builtin_shufflevector(a0, a1,
                     0,1,2,3,4,5,6,7,8,9,10,11,12,13,14,15);
        v16h b = __builtin_shufflevector(b0, b1,
                     0,1,2,3,4,5,6,7,8,9,10,11,12,13,14,15);
        acc = __builtin_amdgcn_wmma_f32_16x16x32_f16(
                  false, a, false, b, (short)0, acc, false, false);
    }

    // ---- store D: lane's 8 f32 are consecutive x (fixed oc, y) ----------
    // W is a multiple of 16, so a wave's 16-pixel tile is all-valid or all-
    // invalid; all addresses 16B aligned.
    int gx0 = x0 + wv * 16 + khalf * 8;
    if (oc < Cout && gx0 < W) {
        float r0 = acc[0], r1 = acc[1], r2 = acc[2], r3 = acc[3];
        float r4 = acc[4], r5 = acc[5], r6 = acc[6], r7 = acc[7];
        if (relu) {
            r0 = fmaxf(r0, 0.f); r1 = fmaxf(r1, 0.f);
            r2 = fmaxf(r2, 0.f); r3 = fmaxf(r3, 0.f);
            r4 = fmaxf(r4, 0.f); r5 = fmaxf(r5, 0.f);
            r6 = fmaxf(r6, 0.f); r7 = fmaxf(r7, 0.f);
        }
        float* dst = out + (long)oc * HW + (long)y * W + gx0;
        *(float4*)dst       = make_float4(r0, r1, r2, r3);
        *(float4*)(dst + 4) = make_float4(r4, r5, r6, r7);
    }
}

template<int CIN>
static void launch_conv(const float* in, const float* w, const float* b, float* out,
                        int Cout, int Hc, int Wc, int relu, hipStream_t stream)
{
    constexpr int K = CIN * 9, Kpad = (K + 31) & ~31, Krow = Kpad + 8;
    size_t lds = (size_t)(64 + 16) * Krow * sizeof(_Float16);
    dim3 grid((Wc + TILE_W - 1) / TILE_W, Hc, (Cout + 15) / 16);
    conv3x3_wmma<CIN><<<grid, dim3(128), lds, stream>>>(in, w, b, out,
                                                        Cout, Hc, Wc, relu);
}

// ---------------------------------------------------------------------------
// 2x2 max pool, 16 channels
// ---------------------------------------------------------------------------
__global__ void maxpool2_kernel(const float* __restrict__ in, float* __restrict__ out,
                                int C, int H, int W)
{
    int Hh = H >> 1, Wh = W >> 1;
    long idx = (long)blockIdx.x * blockDim.x + threadIdx.x;
    long total = (long)C * Hh * Wh;
    if (idx >= total) return;
    int c = (int)(idx / ((long)Hh * Wh));
    int rem = (int)(idx % ((long)Hh * Wh));
    int yy = rem / Wh, xx = rem % Wh;
    const float* p = in + (long)c * H * W + (long)(2 * yy) * W + 2 * xx;
    float a = fmaxf(p[0], p[1]);
    float b = fmaxf(p[W], p[W + 1]);
    out[idx] = fmaxf(a, b);
}

// ---------------------------------------------------------------------------
// 2x2 transposed-conv upsample (einsum bchw,cokl->bohkwl) + bias + skip add
//   h: [16][Hh][Wh], w: [c=16][o=16][2][2], out: [16][2Hh][2Wh]
// ---------------------------------------------------------------------------
__global__ void upsample_add_kernel(const float* __restrict__ h, const float* __restrict__ w,
                                    const float* __restrict__ bias, const float* __restrict__ skip,
                                    float* __restrict__ out, int Hh, int Wh)
{
    long idx = (long)blockIdx.x * blockDim.x + threadIdx.x;
    long total = 16L * Hh * Wh;
    if (idx >= total) return;
    int o = (int)(idx / ((long)Hh * Wh));
    int rem = (int)(idx % ((long)Hh * Wh));
    int yy = rem / Wh, xx = rem % Wh;

    float acc0 = 0.f, acc1 = 0.f, acc2 = 0.f, acc3 = 0.f;
#pragma unroll
    for (int c = 0; c < 16; ++c) {
        float hv = h[(long)c * Hh * Wh + rem];
        const float* wp = w + ((long)c * 16 + o) * 4;
        acc0 += hv * wp[0]; acc1 += hv * wp[1];
        acc2 += hv * wp[2]; acc3 += hv * wp[3];
    }
    int H = 2 * Hh, W = 2 * Wh;
    float bo = bias[o];
    long base = (long)o * H * W + (long)(2 * yy) * W + 2 * xx;
    out[base]         = acc0 + bo + skip[base];
    out[base + 1]     = acc1 + bo + skip[base + 1];
    out[base + W]     = acc2 + bo + skip[base + W];
    out[base + W + 1] = acc3 + bo + skip[base + W + 1];
}

// ---------------------------------------------------------------------------
// Bilateral filter. comb = [x channels 0..3][feats 0..7] (12 ch).
// fp = [cw[12], ax, ay]. Writes 4-channel fout = num/den.
// ---------------------------------------------------------------------------
__global__ void bilateral_kernel(const float* __restrict__ xb, const float* __restrict__ feats,
                                 const float* __restrict__ fp, float* __restrict__ fout,
                                 int H, int W, int ks, int dil)
{
    long idx = (long)blockIdx.x * blockDim.x + threadIdx.x;
    long HW = (long)H * W;
    if (idx >= HW) return;
    int y = (int)(idx / W), x = (int)(idx % W);

    float cw[12];
#pragma unroll
    for (int c = 0; c < 12; ++c) cw[c] = fp[c];
    float ax = fp[12], ay = fp[13];

    float ctr[12];
#pragma unroll
    for (int c = 0; c < 4; ++c) ctr[c] = xb[(long)c * HW + idx];
#pragma unroll
    for (int c = 0; c < 8; ++c) ctr[4 + c] = feats[(long)c * HW + idx];

    float num0 = 0.f, num1 = 0.f, num2 = 0.f, num3 = 0.f, den = 0.f;
    int r = ks >> 1;
    for (int oy = -r; oy <= r; ++oy) {
        int dy = oy * dil, sy = y + dy;
        if (sy < 0 || sy >= H) continue;
        for (int ox = -r; ox <= r; ++ox) {
            int dx = ox * dil, sx = x + dx;
            if (sx < 0 || sx >= W) continue;
            long sp = (long)sy * W + sx;
            float s[12];
#pragma unroll
            for (int c = 0; c < 4; ++c) s[c] = xb[(long)c * HW + sp];
#pragma unroll
            for (int c = 0; c < 8; ++c) s[4 + c] = feats[(long)c * HW + sp];
            float lw = ax * (float)(dx * dx) + ay * (float)(dy * dy);
#pragma unroll
            for (int c = 0; c < 12; ++c) {
                float d = ctr[c] - s[c];
                lw += cw[c] * d * d;
            }
            float wgt = __expf(lw);
            num0 += wgt * s[0]; num1 += wgt * s[1];
            num2 += wgt * s[2]; num3 += wgt * s[3];
            den += wgt;
        }
    }
    float inv = 1.0f / den;
    fout[idx]          = num0 * inv;
    fout[HW + idx]     = num1 * inv;
    fout[2 * HW + idx] = num2 * inv;
    fout[3 * HW + idx] = num3 * inv;
}

// ---------------------------------------------------------------------------
// Build x = [dyn(4) | fixed(9)]. x ch3 (hid) was written by prior conv.
// blend: dyn = alpha*dyn + (1-alpha)*temporal
// ---------------------------------------------------------------------------
__global__ void build_x_kernel(const float* __restrict__ fi, const float* __restrict__ temporal,
                               const float* __restrict__ alpha, float* __restrict__ xb,
                               long HW, int blend)
{
    long p = (long)blockIdx.x * blockDim.x + threadIdx.x;
    if (p >= HW) return;
#pragma unroll
    for (int c = 0; c < 3; ++c) xb[(long)c * HW + p] = fi[(long)c * HW + p];
#pragma unroll
    for (int c = 0; c < 9; ++c) xb[(long)(4 + c) * HW + p] = fi[(long)(3 + c) * HW + p];
    if (blend) {
        float a = alpha[0], b = 1.0f - a;
#pragma unroll
        for (int c = 0; c < 4; ++c) {
            long o = (long)c * HW + p;
            xb[o] = a * xb[o] + b * temporal[o];
        }
    }
}

// out[c] = albedo(c) * x[c],  albedo = x channels 4..6
__global__ void final_out_kernel(const float* __restrict__ xb, float* __restrict__ out, long HW)
{
    long idx = (long)blockIdx.x * blockDim.x + threadIdx.x;
    if (idx >= 3 * HW) return;
    long c = idx / HW, p = idx % HW;
    out[idx] = xb[(4 + c) * HW + p] * xb[c * HW + p];
}

// ---------------------------------------------------------------------------
// Host orchestration (graph-capture safe: only launches + async D2D copies)
// ---------------------------------------------------------------------------
extern "C" void kernel_launch(void* const* d_in, const int* in_sizes, int n_in,
                              void* d_out, int out_size, void* d_ws, size_t ws_size,
                              hipStream_t stream)
{
    (void)in_sizes; (void)n_in; (void)out_size; (void)ws_size;

    const int H = 544, W = 960;
    const long P = (long)H * W;             // 522240
    const int Hh = H / 2, Wh = W / 2;
    const long Ph = (long)Hh * Wh;

    const float* input = (const float*)d_in[0];
    const float* alpha = (const float*)d_in[1];
    const float* hsb_w0 = (const float*)d_in[2];
    const float* hsb_b0 = (const float*)d_in[3];
    const float* hsb_w1 = (const float*)d_in[4];
    const float* hsb_b1 = (const float*)d_in[5];
    const float* hsb_w2 = (const float*)d_in[6];
    const float* hsb_b2 = (const float*)d_in[7];
    const float* d0a_w = (const float*)d_in[8];   const float* d0a_b = (const float*)d_in[9];
    const float* d0b_w = (const float*)d_in[10];  const float* d0b_b = (const float*)d_in[11];
    const float* d0c_w = (const float*)d_in[12];  const float* d0c_b = (const float*)d_in[13];
    const float* e0_w  = (const float*)d_in[14];  const float* e0_b  = (const float*)d_in[15];
    const float* e1_w  = (const float*)d_in[16];  const float* e1_b  = (const float*)d_in[17];
    const float* e2_w  = (const float*)d_in[18];  const float* e2_b  = (const float*)d_in[19];
    const float* u0_w  = (const float*)d_in[20];  const float* u0_b  = (const float*)d_in[21];
    const float* u1_w  = (const float*)d_in[22];  const float* u1_b  = (const float*)d_in[23];
    const float* u2_w  = (const float*)d_in[24];  const float* u2_b  = (const float*)d_in[25];
    const float* ups_w = (const float*)d_in[26];  const float* ups_b = (const float*)d_in[27];
    const float* fpar  = (const float*)d_in[28];

    // Workspace layout (floats). Total = 101*P ~ 211 MB.
    float* Wk       = (float*)d_ws;
    float* xb       = Wk;             // 13P  persistent within stage loop
    float* temporal = Wk + 13 * P;    // 4P   persists across frames
    float* fout     = Wk + 17 * P;    // 4P
    float* arena    = Wk + 21 * P;    // 80P reused scratch
    // hsb phase:
    float* a24 = arena;               // 24P
    float* a48 = arena + 24 * P;      // 48P
    // unet phase (after hsb is done):
    float* skip  = arena;             // 16P
    float* t0    = arena + 16 * P;    // 8P
    float* t1    = arena + 24 * P;    // 8P
    float* half0 = arena + 32 * P;    // 16ch * Ph = 4P
    float* half1 = arena + 36 * P;    // 4P
    float* up    = arena + 40 * P;    // 16P
    float* u0b   = arena + 56 * P;    // 8P
    float* u1b   = arena + 64 * P;    // 8P
    float* feats = arena + 72 * P;    // 8P

    const int KS[5]  = {7, 5, 5, 5, 5};
    const int DIL[5] = {1, 1, 2, 4, 8};
    // per-stage weight strides
    const long S_d0a = 8 * 13 * 9,  S_d0b = 8 * 8 * 9,  S_d0c = 16 * 8 * 9;
    const long S_e   = 16 * 16 * 9, S_u0  = 8 * 16 * 9, S_u12 = 8 * 8 * 9;
    const long S_ups = 16 * 16 * 4;

    const int TPB = 256;

    for (int frame = 0; frame < 2; ++frame) {
        const float* fi = input + (long)frame * 12 * P;

        // ---- hsb: hid written directly into x channel 3 ------------------
        launch_conv<12>(fi,  hsb_w0, hsb_b0, a24,        24, H, W, 1, stream);
        launch_conv<24>(a24, hsb_w1, hsb_b1, a48,        48, H, W, 1, stream);
        launch_conv<48>(a48, hsb_w2, hsb_b2, xb + 3 * P,  1, H, W, 0, stream);

        build_x_kernel<<<(P + TPB - 1) / TPB, TPB, 0, stream>>>(
            fi, temporal, alpha, xb, P, frame > 0 ? 1 : 0);

        for (int s = 0; s < 5; ++s) {
            // encoder U-Net
            launch_conv<13>(xb, d0a_w + s * S_d0a, d0a_b + s * 8,  t0,   8,  H, W, 1, stream);
            launch_conv<8>( t0, d0b_w + s * S_d0b, d0b_b + s * 8,  t1,   8,  H, W, 1, stream);
            launch_conv<8>( t1, d0c_w + s * S_d0c, d0c_b + s * 16, skip, 16, H, W, 0, stream);
            maxpool2_kernel<<<(16 * Ph + TPB - 1) / TPB, TPB, 0, stream>>>(
                skip, half0, 16, H, W);
            launch_conv<16>(half0, e0_w + s * S_e, e0_b + s * 16, half1, 16, Hh, Wh, 1, stream);
            launch_conv<16>(half1, e1_w + s * S_e, e1_b + s * 16, half0, 16, Hh, Wh, 1, stream);
            launch_conv<16>(half0, e2_w + s * S_e, e2_b + s * 16, half1, 16, Hh, Wh, 1, stream);
            upsample_add_kernel<<<(16 * Ph + TPB - 1) / TPB, TPB, 0, stream>>>(
                half1, ups_w + s * S_ups, ups_b + s * 16, skip, up, Hh, Wh);
            launch_conv<16>(up,  u0_w + s * S_u0,  u0_b + s * 8, u0b,   8, H, W, 1, stream);
            launch_conv<8>( u0b, u1_w + s * S_u12, u1_b + s * 8, u1b,   8, H, W, 1, stream);
            launch_conv<8>( u1b, u2_w + s * S_u12, u2_b + s * 8, feats, 8, H, W, 0, stream);

            // bilateral on [x[:4] | feats], result replaces x[:4]
            bilateral_kernel<<<(P + TPB - 1) / TPB, TPB, 0, stream>>>(
                xb, feats, fpar + s * 14, fout, H, W, KS[s], DIL[s]);
            hipMemcpyAsync(xb, fout, (size_t)(4 * P) * sizeof(float),
                           hipMemcpyDeviceToDevice, stream);
        }

        // output = albedo (x ch4..6) * x ch0..2
        final_out_kernel<<<(3 * P + TPB - 1) / TPB, TPB, 0, stream>>>(
            xb, (float*)d_out + (long)frame * 3 * P, P);
        // temporal = x[:4]
        hipMemcpyAsync(temporal, xb, (size_t)(4 * P) * sizeof(float),
                       hipMemcpyDeviceToDevice, stream);
    }
}
